// NNConvClassifier_5660766896564
// MI455X (gfx1250) — matrix operand order repaired and checked by reading, not compiled, not run on hardware
//
#include <hip/hip_runtime.h>
#include <hip/hip_bf16.h>

typedef __attribute__((ext_vector_type(16))) _Float16 v16h;
typedef __attribute__((ext_vector_type(8)))  float    v8f;
typedef __attribute__((ext_vector_type(4)))  _Float16 h4;
typedef __attribute__((ext_vector_type(4)))  float    f4;

#define NN 50000
#define NE 400000
#define HID 32
#define NG 64

// ---------------------------------------------------------------- utilities
// 4-wide f32 -> f16 cast (16B load, 8B store)
__global__ void k_cast_f16(const float* __restrict__ x, _Float16* __restrict__ xh, int n4) {
    int i = blockIdx.x * blockDim.x + threadIdx.x;
    if (i >= n4) return;
    f4 v = ((const f4*)x)[i];
    h4 o;
#pragma unroll
    for (int j = 0; j < 4; ++j) o[j] = (_Float16)v[j];
    ((h4*)xh)[i] = o;
}

__global__ void k_deg(const int* __restrict__ dst, float* __restrict__ deg) {
    int e = blockIdx.x * blockDim.x + threadIdx.x;
    if (e < NE) atomicAdd(&deg[dst[e]], 1.0f);
}

// h[e, 4q..4q+3] = relu(ea[e,:] @ w1 + b1), 4 outputs/thread, f16 out
__global__ void k_edge_h(const float* __restrict__ ea, const float* __restrict__ w1,
                         const float* __restrict__ b1, _Float16* __restrict__ h) {
    int idx = blockIdx.x * blockDim.x + threadIdx.x;
    if (idx >= NE * 8) return;
    const int e = idx >> 3, q = idx & 7;
    const float a0 = ea[e * 2 + 0], a1 = ea[e * 2 + 1];
    const f4 w0 = *(const f4*)(w1 + q * 4);
    const f4 wA = *(const f4*)(w1 + HID + q * 4);
    const f4 bb = *(const f4*)(b1 + q * 4);
    h4 o;
#pragma unroll
    for (int j = 0; j < 4; ++j)
        o[j] = (_Float16)fmaxf(fmaf(a0, w0[j], fmaf(a1, wA[j], bb[j])), 0.0f);
    *(h4*)(h + (size_t)e * 32 + q * 4) = o;
}

// ------------------------------------------------- WMMA: P = xh @ perm(ew2)
// P[v, k*32+o] = sum_i xh[v,i] * ew2[k, i*32+o]    -> P is [NN, 1024] f16
__global__ void k_p_gemm(const _Float16* __restrict__ xh,  // [NN,32]
                         const float* __restrict__ ew2,    // [32,1024]
                         _Float16* __restrict__ P) {       // [NN,1024]
    const int lane = threadIdx.x;
    const int half_sel = lane >> 4;
    const int lsub = lane & 15;
    const int coltile = blockIdx.y;          // 0..63

    // B [32 x 16] slice: B[i][c] = ew2[c>>5, i*32 + (c&31)]
    const int c  = coltile * 16 + lsub;
    const int kq = c >> 5, o = c & 31;
    const int ioff = half_sel * 16;          // lanes16-31 hold K=16..31 of B (contiguous k order, same as A)
    v16h b;
#pragma unroll
    for (int j = 0; j < 16; ++j)
        b[j] = (_Float16)ew2[kq * 1024 + (ioff + j) * 32 + o];

    const int ntiles = NN / 16;              // 3125 (exact)
    for (int t = blockIdx.x; t < ntiles; t += gridDim.x) {
        const int m = t * 16 + lsub;
        const _Float16* row = xh + (size_t)m * 32;
        const int tn = t + gridDim.x;        // prefetch next tile's A rows
        if (tn < ntiles)
            __builtin_prefetch(xh + (size_t)(tn * 16 + lsub) * 32, 0, 0);
        const int off = half_sel * 16;       // lanes16-31 hold K=16..31 of A (matches B)
        v16h a;
#pragma unroll
        for (int j = 0; j < 16; ++j) a[j] = row[off + j];
        v8f cacc = {};
        cacc = __builtin_amdgcn_wmma_f32_16x16x32_f16(
            false, a, false, b, (short)0, cacc, false, false);
        const int ccol  = coltile * 16 + lsub;
        const int mbase = t * 16 + half_sel * 8;   // ISA D layout: vgpr j -> row j(+8)
#pragma unroll
        for (int j = 0; j < 8; ++j)
            P[(size_t)(mbase + j) * 1024 + ccol] = (_Float16)cacc[j];
    }
}

// ---------------------------------- WMMA: out[N,32] = xh[N,32] @ w[32,32] (f32 out)
__global__ void k_gemm_n32(const _Float16* __restrict__ xh, const float* __restrict__ w,
                           float* __restrict__ out) {
    const int lane = threadIdx.x;
    const int half_sel = lane >> 4;
    const int lsub = lane & 15;
    const int coltile = blockIdx.y;          // 0..1

    const int c = coltile * 16 + lsub;       // 0..31
    const int ioff = half_sel * 16;
    v16h b;
#pragma unroll
    for (int j = 0; j < 16; ++j)
        b[j] = (_Float16)w[(ioff + j) * 32 + c];

    const int ntiles = NN / 16;
    for (int t = blockIdx.x; t < ntiles; t += gridDim.x) {
        const int m = t * 16 + lsub;
        const _Float16* row = xh + (size_t)m * 32;
        const int tn = t + gridDim.x;
        if (tn < ntiles)
            __builtin_prefetch(xh + (size_t)(tn * 16 + lsub) * 32, 0, 0);
        const int off = half_sel * 16;
        v16h a;
#pragma unroll
        for (int j = 0; j < 16; ++j) a[j] = row[off + j];
        v8f cacc = {};
        cacc = __builtin_amdgcn_wmma_f32_16x16x32_f16(
            false, a, false, b, (short)0, cacc, false, false);
        const int mbase = t * 16 + half_sel * 8;
#pragma unroll
        for (int j = 0; j < 8; ++j)
            out[(size_t)(mbase + j) * 32 + c] = cacc[j];
    }
}

// ------------------------------------- fused message + scatter-add (4 outputs/thread)
// msg[e,o] = Q[src,o] + sum_k h[e,k] * P[src, k*32+o];  agg[dst,o] += msg
__global__ void k_msg(const int* __restrict__ src, const int* __restrict__ dst,
                      const _Float16* __restrict__ h, const _Float16* __restrict__ P,
                      const float* __restrict__ Q, float* __restrict__ agg) {
    int idx = blockIdx.x * blockDim.x + threadIdx.x;
    if (idx >= NE * 8) return;
    const int e = idx >> 3, q = idx & 7;       // q selects outputs 4q..4q+3
    const int s = src[e], d = dst[e];

    const _Float16* hrow = h + (size_t)e * 32; // uniform across the 8 lanes of this edge
    h4 hv[8];
#pragma unroll
    for (int j = 0; j < 8; ++j) hv[j] = *(const h4*)(hrow + 4 * j);

    const _Float16* pbase = P + (size_t)s * 1024 + q * 4;
    f4 acc = *(const f4*)(Q + (size_t)s * 32 + q * 4);
#pragma unroll
    for (int k = 0; k < 32; ++k) {
        const float hk = (float)hv[k >> 2][k & 3];
        const h4 pv = *(const h4*)(pbase + k * 32);   // 8B load, 64B/edge coalesced
#pragma unroll
        for (int j = 0; j < 4; ++j) acc[j] = fmaf(hk, (float)pv[j], acc[j]);
    }
    float* ap = agg + (size_t)d * 32 + q * 4;
#pragma unroll
    for (int j = 0; j < 4; ++j) atomicAdd(ap + j, acc[j]);
}

// xout = relu(agg/deg + R + bias), 4-wide
__global__ void k_update(const float* __restrict__ agg, const float* __restrict__ deg,
                         const float* __restrict__ R, const float* __restrict__ bias,
                         float* __restrict__ xout) {
    int idx = blockIdx.x * blockDim.x + threadIdx.x;   // NN*8 threads
    if (idx >= NN * 8) return;
    const int v = idx >> 3, q = idx & 7;
    const float dv = 1.0f / fmaxf(deg[v], 1.0f);
    const f4 a = ((const f4*)agg)[idx];
    const f4 r = ((const f4*)R)[idx];
    const f4 bb = *(const f4*)(bias + q * 4);
    f4 o;
#pragma unroll
    for (int j = 0; j < 4; ++j) o[j] = fmaxf(fmaf(a[j], dv, r[j] + bb[j]), 0.0f);
    ((f4*)xout)[idx] = o;
}

// ------------------------------------- pooling + classifier
__global__ void k_pool(const float* __restrict__ x, const int* __restrict__ batch,
                       float* __restrict__ pooled, float* __restrict__ cnt) {
    int idx = blockIdx.x * blockDim.x + threadIdx.x;   // NN*8 threads
    if (idx >= NN * 8) return;
    const int v = idx >> 3, q = idx & 7;
    const int g = batch[v];
    const f4 xv = ((const f4*)x)[idx];
    float* pp = pooled + (size_t)g * HID + q * 4;
#pragma unroll
    for (int j = 0; j < 4; ++j) atomicAdd(pp + j, xv[j]);
    if (q == 0) atomicAdd(&cnt[g], 1.0f);
}

__global__ void k_cls(const float* __restrict__ pooled, const float* __restrict__ cnt,
                      const float* __restrict__ cw1, const float* __restrict__ cb1,
                      const float* __restrict__ cw2, const float* __restrict__ cb2,
                      float* __restrict__ out) {
    int g = threadIdx.x;
    if (g >= NG) return;
    float inv = 1.0f / fmaxf(cnt[g], 1.0f);
    float p[HID];
#pragma unroll
    for (int c2 = 0; c2 < HID; ++c2) p[c2] = pooled[g * HID + c2] * inv;
    float hb[16];
#pragma unroll
    for (int j = 0; j < 16; ++j) {
        float a = cb1[j];
        for (int c2 = 0; c2 < HID; ++c2) a = fmaf(p[c2], cw1[c2 * 16 + j], a);
        hb[j] = fmaxf(a, 0.0f);
    }
#pragma unroll
    for (int t = 0; t < 2; ++t) {
        float a = cb2[t];
        for (int j = 0; j < 16; ++j) a = fmaf(hb[j], cw2[j * 2 + t], a);
        out[g * 2 + t] = a;
    }
}

// ---------------------------------------------------------------- launcher
extern "C" void kernel_launch(void* const* d_in, const int* in_sizes, int n_in,
                              void* d_out, int out_size, void* d_ws, size_t ws_size,
                              hipStream_t stream) {
    const float* x     = (const float*)d_in[0];
    const int*   eidx  = (const int*)d_in[1];
    const float* eattr = (const float*)d_in[2];
    const int*   batch = (const int*)d_in[3];
    const int* src = eidx;
    const int* dst = eidx + NE;
    const float* cw1 = (const float*)d_in[22];
    const float* cb1 = (const float*)d_in[23];
    const float* cw2 = (const float*)d_in[24];
    const float* cb2 = (const float*)d_in[25];

    // workspace carve-out (256B aligned)
    char* ws = (char*)d_ws;
    size_t off = 0;
    auto carve = [&](size_t bytes) -> char* {
        char* p = ws + off;
        off = (off + bytes + 255) & ~(size_t)255;
        return p;
    };
    _Float16* xh  = (_Float16*)carve((size_t)NN * 32 * 2);
    _Float16* hbuf= (_Float16*)carve((size_t)NE * 32 * 2);
    _Float16* P   = (_Float16*)carve((size_t)NN * 1024 * 2);   // 102 MB: L2-resident
    float* Q      = (float*)carve((size_t)NN * 32 * 4);
    float* R      = (float*)carve((size_t)NN * 32 * 4);
    float* agg    = (float*)carve((size_t)NN * 32 * 4);
    float* deg    = (float*)carve((size_t)NN * 4);
    float* xb0    = (float*)carve((size_t)NN * 32 * 4);
    float* xb1    = (float*)carve((size_t)NN * 32 * 4);
    float* pooled = (float*)carve((size_t)NG * 32 * 4);
    float* cnt    = (float*)carve((size_t)NG * 4);

    const int B = 256;
    const dim3 gP(392, 64);   // 3125 node tiles striped over 392 blocks x 64 col tiles
    const dim3 gR(392, 2);

    hipMemsetAsync(deg, 0, (size_t)NN * 4, stream);
    k_deg<<<(NE + B - 1) / B, B, 0, stream>>>(dst, deg);

    const float* cur = x;
    float* nxt = xb0;
    for (int l = 0; l < 3; ++l) {
        const float* ew1  = (const float*)d_in[4 + l * 6 + 0];
        const float* eb1  = (const float*)d_in[4 + l * 6 + 1];
        const float* ew2  = (const float*)d_in[4 + l * 6 + 2];
        const float* eb2  = (const float*)d_in[4 + l * 6 + 3];
        const float* root = (const float*)d_in[4 + l * 6 + 4];
        const float* bias = (const float*)d_in[4 + l * 6 + 5];

        k_cast_f16<<<(NN * 8 + B - 1) / B, B, 0, stream>>>(cur, xh, NN * 8);
        k_edge_h<<<(NE * 8 + B - 1) / B, B, 0, stream>>>(eattr, ew1, eb1, hbuf);
        k_p_gemm<<<gP, 32, 0, stream>>>(xh, ew2, P);
        k_gemm_n32<<<gR, 32, 0, stream>>>(xh, root, R);
        k_gemm_n32<<<gR, 32, 0, stream>>>(xh, eb2, Q);   // eb2 viewed as [32,32]
        hipMemsetAsync(agg, 0, (size_t)NN * 32 * 4, stream);
        k_msg<<<(NE * 8 + B - 1) / B, B, 0, stream>>>(src, dst, hbuf, P, Q, agg);
        k_update<<<(NN * 8 + B - 1) / B, B, 0, stream>>>(agg, deg, R, bias, nxt);

        cur = nxt;
        nxt = (l == 0) ? xb1 : (float*)((cur == xb0) ? xb1 : xb0);
    }

    hipMemsetAsync(pooled, 0, (size_t)NG * 32 * 4, stream);
    hipMemsetAsync(cnt, 0, (size_t)NG * 4, stream);
    k_pool<<<(NN * 8 + B - 1) / B, B, 0, stream>>>(cur, batch, pooled, cnt);
    k_cls<<<1, 64, 0, stream>>>(pooled, cnt, cw1, cb1, cw2, cb2, (float*)d_out);
}
